// GCRN_13185549599089
// MI455X (gfx1250) — matrix operand order repaired and checked by reading, not compiled, not run on hardware
//
#include <hip/hip_runtime.h>
#include <math.h>

// ---------------------------------------------------------------------------
// Problem constants (from reference): B=2, T=24, N=20000, F_IN=32, H=128,
// E=320000, K=2, OUT=1.
// ---------------------------------------------------------------------------
constexpr int BATCH  = 2;
constexpr int TSTEPS = 24;
constexpr int NNODE  = 20000;
constexpr int FIN    = 32;
constexpr int HDIM   = 128;
constexpr int NEDGE  = 320000;
constexpr int MROWS  = BATCH * NNODE;     // 40000 (divisible by 16)

typedef __attribute__((ext_vector_type(16))) __bf16 v16bf;
typedef __attribute__((ext_vector_type(8)))  __bf16 v8bf;
typedef __attribute__((ext_vector_type(8)))  float  v8f;

__device__ inline __bf16 f2bf(float f) {
  // round-to-nearest-even f32 -> bf16 via bit trick (no reliance on cast lowering)
  unsigned u = __float_as_uint(f);
  unsigned r = u + 0x7FFFu + ((u >> 16) & 1u);
  unsigned short hs = (unsigned short)(r >> 16);
  __bf16 b;
  __builtin_memcpy(&b, &hs, sizeof(b));
  return b;
}

// ---------------------------------------------------------------------------
// Utility kernels
// ---------------------------------------------------------------------------
__global__ void k_zero(float* __restrict__ p, long long n) {
  long long i = (long long)blockIdx.x * blockDim.x + threadIdx.x;
  if (i < n) p[i] = 0.0f;
}

__global__ void k_deg(const long long* __restrict__ src,
                      const float* __restrict__ ew,
                      float* __restrict__ deg) {
  int e = blockIdx.x * blockDim.x + threadIdx.x;
  if (e < NEDGE) atomicAdd(&deg[(int)src[e]], ew[e]);
}

__global__ void k_norm(const long long* __restrict__ src,
                       const long long* __restrict__ dst,
                       const float* __restrict__ ew,
                       const float* __restrict__ deg,
                       float* __restrict__ nrm) {
  int e = blockIdx.x * blockDim.x + threadIdx.x;
  if (e >= NEDGE) return;
  float ds = deg[(int)src[e]];
  float dd = deg[(int)dst[e]];
  float is = ds > 0.0f ? rsqrtf(ds) : 0.0f;
  float id = dd > 0.0f ? rsqrtf(dd) : 0.0f;
  nrm[e] = -is * ew[e] * id;
}

// Build concatenated x-side weights: Wcat[64,384]; k<32 <-> W[0], k>=32 <-> W[1].
// Source arrays are [2,32,128] flat, so linear index is simply k*128+c.
__global__ void k_build_wxcat(const float* __restrict__ Wxz,
                              const float* __restrict__ Wxr,
                              const float* __restrict__ Wxh,
                              float* __restrict__ Wcat) {
  int i = blockIdx.x * blockDim.x + threadIdx.x;
  const int total = 64 * 384;
  if (i >= total) return;
  int k = i / 384, n = i % 384;
  int g = n >> 7, c = n & 127;
  const float* W = (g == 0) ? Wxz : ((g == 1) ? Wxr : Wxh);
  Wcat[i] = W[k * 128 + c];
}

// Build concatenated h-side weights: Wcat[256, ngates*128]
__global__ void k_build_whcat(const float* __restrict__ W0,
                              const float* __restrict__ W1,
                              float* __restrict__ Wcat, int ngates) {
  int ncol = ngates * 128;
  long long total = (long long)256 * ncol;
  long long i = (long long)blockIdx.x * blockDim.x + threadIdx.x;
  if (i >= total) return;
  int k = (int)(i / ncol), n = (int)(i % ncol);
  int g = n >> 7, c = n & 127;
  const float* W = g ? W1 : W0;
  Wcat[i] = W[k * 128 + c];
}

// Pack W[Kdim,Ncol] (f32 row-major) into WMMA B-fragment lane order (bf16):
// frag[((nt*KT+kt)*32 + lane)*16 + j] = W[kt*32 + (lane>=16?8:0) + (j<8 ? j : j+8)][nt*16 + (lane&15)]
__global__ void k_pack_frag(const float* __restrict__ W, __bf16* __restrict__ frag,
                            int Kdim, int Ncol) {
  long long total = (long long)Kdim * Ncol;
  long long i = (long long)blockIdx.x * blockDim.x + threadIdx.x;
  if (i >= total) return;
  int KT = Kdim >> 5;
  int j = (int)(i & 15);
  int lane = (int)((i >> 4) & 31);
  int kt = (int)((i >> 9) % KT);
  int nt = (int)((i >> 9) / KT);
  int n = (nt << 4) + (lane & 15);
  int k = (kt << 5) + ((lane >= 16) ? 8 : 0) + ((j < 8) ? j : j + 8);
  frag[i] = f2bf(W[(long long)k * Ncol + n]);
}

// Edge scatter: out[b, dst, :] += norm[e] * x[b, src, :]   (float4 + f32 atomics)
// For C=128 one wave32 covers exactly one edge (32 lanes x float4).
template <int C>
__global__ void k_scatter(const long long* __restrict__ src,
                          const long long* __restrict__ dst,
                          const float* __restrict__ nrm,
                          const float* __restrict__ xbase, long long xBatchStride,
                          float* __restrict__ out, long long oBatchStride) {
  constexpr int PER = C / 4;
  long long i = (long long)blockIdx.x * blockDim.x + threadIdx.x;
  long long total = (long long)NEDGE * PER;
  if (i >= total) return;
  int e  = (int)(i / PER);
  int c4 = (int)(i % PER) * 4;
  int b  = blockIdx.y;
  int s  = (int)src[e];
  int d  = (int)dst[e];
  float w = nrm[e];
  const float4 v = *(const float4*)(xbase + (long long)b * xBatchStride + (long long)s * C + c4);
  float* o = out + (long long)b * oBatchStride + (long long)d * C + c4;
  atomicAdd(o + 0, w * v.x);
  atomicAdd(o + 1, w * v.y);
  atomicAdd(o + 2, w * v.z);
  atomicAdd(o + 3, w * v.w);
}

// Build bf16 A = [x_t | prop_x]  ([M,64])
__global__ void k_acat_x(const float* __restrict__ xt /* x_seq + t*N*FIN */,
                         const float* __restrict__ px,
                         __bf16* __restrict__ dstp) {
  long long i = (long long)blockIdx.x * blockDim.x + threadIdx.x;
  long long total = (long long)MROWS * 64;
  if (i >= total) return;
  int m = (int)(i >> 6), k = (int)(i & 63);
  float v;
  if (k < 32) {
    int b = m / NNODE, n = m % NNODE;
    v = xt[(long long)b * ((long long)TSTEPS * NNODE * FIN) + (long long)n * FIN + k];
  } else {
    v = px[(long long)m * 32 + (k - 32)];
  }
  dstp[i] = f2bf(v);
}

// Build bf16 A = [h | prop_h]  ([M,256])
__global__ void k_acat_h(const float* __restrict__ h,
                         const float* __restrict__ ph,
                         __bf16* __restrict__ dstp) {
  long long i = (long long)blockIdx.x * blockDim.x + threadIdx.x;
  long long total = (long long)MROWS * 256;
  if (i >= total) return;
  int m = (int)(i >> 8), k = (int)(i & 255);
  float v = (k < 128) ? h[(long long)m * 128 + k] : ph[(long long)m * 128 + (k - 128)];
  dstp[i] = f2bf(v);
}

// ---------------------------------------------------------------------------
// WMMA GEMM: C[M,NCOL] = A[M,KDIM](bf16) @ Bfrag(KDIM,NCOL)(bf16) with fused
// epilogues. Each wave owns a 16x64 output strip: 4 N-subtiles = 4 independent
// accumulator chains (no WMMA->WMMA RAW hazard back-to-back) and 4x reuse of
// each A fragment load.
//   MODE 0: store raw f32 (Gx, NCOL=384)
//   MODE 1: z/r gates: pre = acc + Gx[:,col] + bias; sigmoid;
//           col<128 -> zbuf, col>=128 -> hr = h * r           (NCOL=256)
//   MODE 2: h~ = tanh(acc + Gx[:,256+col] + bias); h = z*h + (1-z)*h~ (NCOL=128)
// Tile guards are wave-uniform (EXEC stays all-1s for WMMA).
// ---------------------------------------------------------------------------
template <int MODE, int KDIM, int NCOL>
__global__ __launch_bounds__(256) void k_gemm(
    const __bf16* __restrict__ A, const __bf16* __restrict__ Bfrag,
    const float* __restrict__ Gx,
    const float* __restrict__ b0x, const float* __restrict__ b0h,
    const float* __restrict__ b1x, const float* __restrict__ b1h,
    const float* __restrict__ hbuf, const float* __restrict__ zbuf,
    float* __restrict__ out0, float* __restrict__ out1) {
  constexpr int KT = KDIM / 32;     // K-steps of 32
  constexpr int NSUB = 4;           // 16-col subtiles per wave
  constexpr int NGRP = NCOL / (16 * NSUB);
  constexpr int MT = MROWS / 16;
  int wave = blockIdx.x * (blockDim.x >> 5) + (threadIdx.x >> 5);
  if (wave >= MT * NGRP) return;                 // wave-uniform exit
  int lane = threadIdx.x & 31;
  int ng = wave % NGRP, mt = wave / NGRP;
  int m0 = mt << 4;

  // A fragment addressing (16-bit A 16x32 layout, ISA 7.12.2):
  // lane<16: row m0+lane, K {kt*32+0..7, kt*32+16..23}; lane>=16: K {8..15, 24..31}
  int arow  = m0 + (lane & 15);
  int kbase = (lane >> 4) * 8;
  const __bf16* arp = A + (long long)arow * KDIM + kbase;
  const __bf16* brp = Bfrag + ((long long)(ng * NSUB * KT) * 32 + lane) * 16;

  v8f acc[NSUB];
#pragma unroll
  for (int s = 0; s < NSUB; ++s) acc[s] = v8f{0.f, 0.f, 0.f, 0.f, 0.f, 0.f, 0.f, 0.f};

#pragma unroll
  for (int kt = 0; kt < KT; ++kt) {
    v8bf lo = *(const v8bf*)(arp + kt * 32);
    v8bf hi = *(const v8bf*)(arp + kt * 32 + 16);
    v16bf a;
#pragma unroll
    for (int q = 0; q < 8; ++q) { a[q] = lo[q]; a[q + 8] = hi[q]; }
#pragma unroll
    for (int s = 0; s < NSUB; ++s) {
      // subtile s uses fragment block (ng*NSUB + s)*KT + kt
      v16bf b = *(const v16bf*)(brp + ((long long)s * KT + kt) * 512);
      acc[s] = __builtin_amdgcn_wmma_f32_16x16x32_bf16(
          /*neg_a=*/false, a, /*neg_b=*/false, b,
          /*c_mod=*/(short)0, acc[s], /*reuse_a=*/false, /*reuse_b=*/false);
    }
  }

  // C/D layout: VGPR j -> row m0 + j (+8 for lanes 16..31), col = nt*16 + (lane&15)
  int mbase = m0 + ((lane >> 4) << 3);
#pragma unroll
  for (int s = 0; s < NSUB; ++s) {
    int col = (ng * NSUB + s) * 16 + (lane & 15);
#pragma unroll
    for (int j = 0; j < 8; ++j) {
      int m = mbase + j;
      float v = acc[s][j];
      if constexpr (MODE == 0) {
        out0[(long long)m * NCOL + col] = v;
      } else if constexpr (MODE == 1) {
        int gate = col >> 7;
        int c = col & 127;
        float bias = gate ? (b1x[c] + b1h[c]) : (b0x[c] + b0h[c]);
        float pre = v + Gx[(long long)m * 384 + col] + bias;
        float sg = 1.0f / (1.0f + __expf(-pre));
        if (gate == 0) out0[(long long)m * 128 + c] = sg;                      // z
        else           out1[(long long)m * 128 + c] = hbuf[(long long)m * 128 + c] * sg; // h*r
      } else {
        int c = col;
        float pre = v + Gx[(long long)m * 384 + 256 + c] + b0x[c] + b0h[c];
        float th = tanhf(pre);
        float zv = zbuf[(long long)m * 128 + c];
        float hold = hbuf[(long long)m * 128 + c];
        out0[(long long)m * 128 + c] = zv * hold + (1.0f - zv) * th;          // h_new
      }
    }
  }
}

// Final readout: out[m] = relu(h[m,:]) . Wl + bl   (one wave32 per row)
__global__ void k_final(const float* __restrict__ h, const float* __restrict__ Wl,
                        const float* __restrict__ bl, float* __restrict__ out) {
  int wave = (blockIdx.x * blockDim.x + threadIdx.x) >> 5;
  int lane = threadIdx.x & 31;
  if (wave >= MROWS) return;
  float acc = 0.0f;
#pragma unroll
  for (int c = lane; c < HDIM; c += 32) {
    float hv = h[(long long)wave * HDIM + c];
    hv = hv > 0.0f ? hv : 0.0f;
    acc += hv * Wl[c];
  }
#pragma unroll
  for (int off = 16; off > 0; off >>= 1) acc += __shfl_down(acc, off, 32);
  if (lane == 0) out[wave] = acc + bl[0];
}

// ---------------------------------------------------------------------------
// Host launcher
// ---------------------------------------------------------------------------
static inline dim3 g1(long long n, int bs) { return dim3((unsigned)((n + bs - 1) / bs)); }

extern "C" void kernel_launch(void* const* d_in, const int* in_sizes, int n_in,
                              void* d_out, int out_size, void* d_ws, size_t ws_size,
                              hipStream_t stream) {
  const float*     x_seq = (const float*)d_in[0];
  const long long* eidx  = (const long long*)d_in[1];
  const long long* esrc  = eidx;
  const long long* edst  = eidx + NEDGE;
  const float* ew  = (const float*)d_in[2];
  const float* Wxz = (const float*)d_in[3];  const float* bxz = (const float*)d_in[4];
  const float* Whz = (const float*)d_in[5];  const float* bhz = (const float*)d_in[6];
  const float* Wxr = (const float*)d_in[7];  const float* bxr = (const float*)d_in[8];
  const float* Whr = (const float*)d_in[9];  const float* bhr = (const float*)d_in[10];
  const float* Wxh = (const float*)d_in[11]; const float* bxh = (const float*)d_in[12];
  const float* Whh = (const float*)d_in[13]; const float* bhh = (const float*)d_in[14];
  const float* Wl  = (const float*)d_in[15]; const float* bl  = (const float*)d_in[16];

  // Workspace carve-out (256B aligned)
  char* base = (char*)d_ws;
  size_t off = 0;
  auto alloc = [&](size_t bytes) -> void* {
    void* p = base + off;
    off = (off + bytes + 255) & ~(size_t)255;
    return p;
  };
  float*  deg      = (float*)alloc((size_t)NNODE * 4);
  float*  nrm      = (float*)alloc((size_t)NEDGE * 4);
  float*  wxcat    = (float*)alloc((size_t)64 * 384 * 4);
  float*  whzrcat  = (float*)alloc((size_t)256 * 256 * 4);
  float*  whhcat   = (float*)alloc((size_t)256 * 128 * 4);
  __bf16* wxfrag   = (__bf16*)alloc((size_t)64 * 384 * 2);
  __bf16* whzrfrag = (__bf16*)alloc((size_t)256 * 256 * 2);
  __bf16* whhfrag  = (__bf16*)alloc((size_t)256 * 128 * 2);
  float*  px   = (float*)alloc((size_t)MROWS * 32 * 4);
  __bf16* acx  = (__bf16*)alloc((size_t)MROWS * 64 * 2);
  float*  Gx   = (float*)alloc((size_t)MROWS * 384 * 4);
  float*  ph   = (float*)alloc((size_t)MROWS * 128 * 4);
  __bf16* ach  = (__bf16*)alloc((size_t)MROWS * 256 * 2);
  float*  zb   = (float*)alloc((size_t)MROWS * 128 * 4);
  float*  hr   = (float*)alloc((size_t)MROWS * 128 * 4);
  float*  phr  = (float*)alloc((size_t)MROWS * 128 * 4);
  __bf16* achr = (__bf16*)alloc((size_t)MROWS * 256 * 2);
  float*  hbuf = (float*)alloc((size_t)MROWS * 128 * 4);

  const int BS = 256;

  // ---- Graph normalization + weight packing (cheap, every call; deterministic)
  k_zero<<<g1(NNODE, BS), BS, 0, stream>>>(deg, NNODE);
  k_deg<<<g1(NEDGE, BS), BS, 0, stream>>>(esrc, ew, deg);
  k_norm<<<g1(NEDGE, BS), BS, 0, stream>>>(esrc, edst, ew, deg, nrm);

  k_build_wxcat<<<g1(64 * 384, BS), BS, 0, stream>>>(Wxz, Wxr, Wxh, wxcat);
  k_build_whcat<<<g1((long long)256 * 256, BS), BS, 0, stream>>>(Whz, Whr, whzrcat, 2);
  k_build_whcat<<<g1((long long)256 * 128, BS), BS, 0, stream>>>(Whh, Whh, whhcat, 1);
  k_pack_frag<<<g1((long long)64 * 384, BS), BS, 0, stream>>>(wxcat, wxfrag, 64, 384);
  k_pack_frag<<<g1((long long)256 * 256, BS), BS, 0, stream>>>(whzrcat, whzrfrag, 256, 256);
  k_pack_frag<<<g1((long long)256 * 128, BS), BS, 0, stream>>>(whhcat, whhfrag, 256, 128);

  k_zero<<<g1((long long)MROWS * 128, BS), BS, 0, stream>>>(hbuf, (long long)MROWS * 128);

  const long long xBStride = (long long)TSTEPS * NNODE * FIN;  // batch stride in x_seq
  const long long h32  = (long long)NNODE * 32;                // batch stride [N,32]
  const long long h128 = (long long)NNODE * 128;               // batch stride [N,128]

  for (int t = 0; t < TSTEPS; ++t) {
    const float* xt = x_seq + (size_t)t * NNODE * FIN;

    // prop_x = Ltilde @ x_t
    k_zero<<<g1((long long)MROWS * 32, BS), BS, 0, stream>>>(px, (long long)MROWS * 32);
    {
      dim3 gs((unsigned)(((long long)NEDGE * 8 + BS - 1) / BS), BATCH);
      k_scatter<32><<<gs, BS, 0, stream>>>(esrc, edst, nrm, xt, xBStride, px, h32);
    }
    k_acat_x<<<g1((long long)MROWS * 64, BS), BS, 0, stream>>>(xt, px, acx);

    // Gx[M,384] = [x|prop_x] @ Wxcat   (z|r|h x-side contributions)
    {
      long long tiles = (MROWS / 16) * (384 / 64);
      k_gemm<0, 64, 384><<<g1((tiles + 7) / 8 * 8LL * 32, BS), BS, 0, stream>>>(
          acx, wxfrag, nullptr, nullptr, nullptr, nullptr, nullptr, nullptr, nullptr, Gx, nullptr);
    }

    // prop_h = Ltilde @ h
    k_zero<<<g1((long long)MROWS * 128, BS), BS, 0, stream>>>(ph, (long long)MROWS * 128);
    {
      dim3 gs((unsigned)(((long long)NEDGE * 32 + BS - 1) / BS), BATCH);
      k_scatter<128><<<gs, BS, 0, stream>>>(esrc, edst, nrm, hbuf, h128, ph, h128);
    }
    k_acat_h<<<g1((long long)MROWS * 256, BS), BS, 0, stream>>>(hbuf, ph, ach);

    // z = sigmoid(...), hr = h*r = h*sigmoid(...)
    {
      long long tiles = (MROWS / 16) * (256 / 64);
      k_gemm<1, 256, 256><<<g1((tiles + 7) / 8 * 8LL * 32, BS), BS, 0, stream>>>(
          ach, whzrfrag, Gx, bxz, bhz, bxr, bhr, hbuf, nullptr, zb, hr);
    }

    // prop_hr = Ltilde @ (h*r)
    k_zero<<<g1((long long)MROWS * 128, BS), BS, 0, stream>>>(phr, (long long)MROWS * 128);
    {
      dim3 gs((unsigned)(((long long)NEDGE * 32 + BS - 1) / BS), BATCH);
      k_scatter<128><<<gs, BS, 0, stream>>>(esrc, edst, nrm, hr, h128, phr, h128);
    }
    k_acat_h<<<g1((long long)MROWS * 256, BS), BS, 0, stream>>>(hr, phr, achr);

    // h = z*h + (1-z)*tanh(...)   (in-place update; each element owned by one lane)
    {
      long long tiles = (MROWS / 16) * (128 / 64);
      k_gemm<2, 256, 128><<<g1((tiles + 7) / 8 * 8LL * 32, BS), BS, 0, stream>>>(
          achr, whhfrag, Gx, bxh, bhh, nullptr, nullptr, hbuf, zb, hbuf, nullptr);
    }
  }

  // out[B,N,1] = relu(h) @ Wl + bl
  k_final<<<g1((long long)MROWS * 32, BS), BS, 0, stream>>>(hbuf, Wl, bl, (float*)d_out);
}
